// MultiHeadAttention_42923903156587
// MI455X (gfx1250) — matrix operand order repaired and checked by reading, not compiled, not run on hardware
//
#include <hip/hip_runtime.h>
#include <hip/hip_bf16.h>

// Problem constants
#define B_   2
#define H_   16
#define CK_  64
#define T_   2048
#define C_   1024
#define COUT_ 1024

typedef __attribute__((ext_vector_type(16))) __bf16 v16bf;
typedef __attribute__((ext_vector_type(8)))  __bf16 v8bf;
typedef __attribute__((ext_vector_type(8)))  float  v8f;
typedef __attribute__((ext_vector_type(4)))  unsigned int v4u;
typedef __attribute__((ext_vector_type(8)))  int v8i;
typedef __attribute__((ext_vector_type(4)))  int v4i;

// Native convert: compiler emits v_cvt_pk_bf16_f32 pairs.
__device__ __forceinline__ __bf16 f2bf(float f) { return (__bf16)f; }

__device__ __forceinline__ v8f zero8() {
  v8f z;
#pragma unroll
  for (int i = 0; i < 8; ++i) z[i] = 0.f;
  return z;
}

__device__ __forceinline__ v16bf cat8(v8bf lo, v8bf hi) {
  v16bf r;
#pragma unroll
  for (int i = 0; i < 8; ++i) { r[i] = lo[i]; r[i + 8] = hi[i]; }
  return r;
}

// A-operand element->K mapping for 16-bit 16x32 A matrix (ISA 7.12.2)
__device__ __forceinline__ int amapK(int e, int half) {
  return e + (half ? 8 : 0) + (e >= 8 ? 8 : 0);
}

__device__ __forceinline__ v8f wmma_bf16(v16bf a, v16bf b, v8f c) {
  return __builtin_amdgcn_wmma_f32_16x16x32_bf16(false, a, false, b, (short)0, c,
                                                 false, false);
}

__device__ __forceinline__ void async_b128(unsigned int ldsOff,
                                           const void* gptr) {
  asm volatile("global_load_async_to_lds_b128 %0, %1, off" ::"v"(ldsOff),
               "v"((unsigned long long)(size_t)gptr)
               : "memory");
}
__device__ __forceinline__ void async_b128_off16(unsigned int ldsOff,
                                                 const void* gptr) {
  // IOFFSET is added to BOTH the LDS and the global address (ISA 10.7)
  asm volatile("global_load_async_to_lds_b128 %0, %1, off offset:16" ::"v"(
                   ldsOff),
               "v"((unsigned long long)(size_t)gptr)
               : "memory");
}
__device__ __forceinline__ void wait_async0() {
  asm volatile("s_wait_asynccnt 0" ::: "memory");
}

// ---------------------------------------------------------------------------
// TDM: load a 2D tile (tileK rows x tileN cols, 2-byte elems) from a
// row-major (rows x rowStride) bf16 tensor into LDS (packed row-major).
// D# per ISA 8.3/8.4: group0 = {count, lds_addr, global_addr, type=2},
// group1 = {data_size=1(2B), tensor dims/strides, tile dims}.
// ---------------------------------------------------------------------------
__device__ __forceinline__ void tdm_load_tile_2d(unsigned int ldsAddr,
                                                 const void* gptr) {
  unsigned long long ga = (unsigned long long)(size_t)gptr;
  v4u g0;
  g0.x = 1u;                                             // count=1 (valid)
  g0.y = ldsAddr;                                        // lds_addr (bytes)
  g0.z = (unsigned int)ga;                               // global_addr[31:0]
  g0.w = (unsigned int)((ga >> 32) & 0x01FFFFFFull) | (2u << 30);  // type=2
  v8i g1;
  g1[0] = 0x10000;                 // workgroup_mask=0, data_size=1 (2 bytes)
  g1[1] = (int)((T_ & 0xFFFFu) << 16);                   // tensor_dim0 lo16
  g1[2] = (int)(((T_ >> 16) & 0xFFFFu) | ((C_ & 0xFFFFu) << 16));  // d0hi,d1lo
  g1[3] = (int)(((C_ >> 16) & 0xFFFFu) | (16u << 16));   // d1hi, tile_dim0=16
  g1[4] = 32;                                            // tile_dim1=32
  g1[5] = T_;                                            // dim0_stride lo32
  g1[6] = 0;                                             // stride0 hi, stride1 lo
  g1[7] = 0;
  v4i z4 = {0, 0, 0, 0};
#if __clang_major__ >= 23
  v8i z8 = {0, 0, 0, 0, 0, 0, 0, 0};
  __builtin_amdgcn_tensor_load_to_lds(g0, g1, z4, z4, z8, 0);
#else
  __builtin_amdgcn_tensor_load_to_lds(g0, g1, z4, z4, 0);
#endif
}

// ---------------------------------------------------------------------------
// Kernel 0: convert wo (COUT x C fp32) -> bf16
// ---------------------------------------------------------------------------
__global__ __launch_bounds__(256) void cvt_wo_kernel(const float* __restrict__ wo,
                                                     __bf16* __restrict__ wo_bf) {
  int i = blockIdx.x * 256 + threadIdx.x;
  wo_bf[i] = f2bf(wo[i]);
}

// ---------------------------------------------------------------------------
// Kernel 1: QKV projection (WMMA bf16) + bias + RoPE.
// grid (B*H, T/16), block 128 (4 waves). Wave w owns d-rows [16w, 16w+16).
// x/c tiles staged global->LDS with CDNA5 async-to-LDS (coalesced 64B rows).
// Outputs: qT/kT bf16 in (bh, t, d) layout, v bf16 in (bh, d, t) layout.
// ---------------------------------------------------------------------------
__global__ __launch_bounds__(128) void qkv_rope_kernel(
    const float* __restrict__ x, const float* __restrict__ cc,
    const float* __restrict__ wq, const float* __restrict__ bq,
    const float* __restrict__ wk, const float* __restrict__ bk,
    const float* __restrict__ wv, const float* __restrict__ bv,
    __bf16* __restrict__ qT, __bf16* __restrict__ kT, __bf16* __restrict__ vW) {
  __shared__ float xs[CK_ * 16];  // [channel][t]
  __shared__ float cs[CK_ * 16];
  __shared__ float qs[CK_ * 16];
  __shared__ float ks[CK_ * 16];

  const int bh = blockIdx.x;
  const int b = bh >> 4, h = bh & 15;
  const int t0 = blockIdx.y * 16;
  const int tid = threadIdx.x;
  const int lane = tid & 31;
  const int w = tid >> 5;
  const int half = lane >> 4;
  const int lm = lane & 15;

  // ---- async stage x/c tiles: 64 rows x 64B. Thread moves 32B of one row.
  const int kr = tid >> 1;   // 0..63 channel row
  const int pr = tid & 1;    // 0/1 -> 32B chunk
  {
    unsigned int lx = (unsigned int)(size_t)(&xs[0]) +
                      (unsigned int)(kr * 16 + pr * 8) * 4u;
    unsigned int lc = (unsigned int)(size_t)(&cs[0]) +
                      (unsigned int)(kr * 16 + pr * 8) * 4u;
    const float* gx = x + ((size_t)b * C_ + h * 64 + kr) * T_ + t0 + pr * 8;
    const float* gc = cc + ((size_t)b * C_ + h * 64 + kr) * T_ + t0 + pr * 8;
    async_b128(lx, gx);
    async_b128_off16(lx, gx);
    async_b128(lc, gc);
    async_b128_off16(lc, gc);
  }

  // ---- A operands: weight rows (d = 16w + lm); small, L2-resident gathers
  const int drow = w * 16 + lm;
  const float* wqp = wq + (size_t)(h * 64 + drow) * 64;
  const float* wkp = wk + (size_t)(h * 64 + drow) * 64;
  const float* wvp = wv + (size_t)(h * 64 + drow) * 64;
  v16bf aq0, aq1, ak0, ak1, av0, av1;
#pragma unroll
  for (int e = 0; e < 16; ++e) {
    int K = amapK(e, half);
    aq0[e] = f2bf(wqp[K]);      aq1[e] = f2bf(wqp[K + 32]);
    ak0[e] = f2bf(wkp[K]);      ak1[e] = f2bf(wkp[K + 32]);
    av0[e] = f2bf(wvp[K]);      av1[e] = f2bf(wvp[K + 32]);
  }

  wait_async0();
  __syncthreads();

  // ---- B operands from LDS (K = channel, N = t)
  v16bf bx0, bx1, bc0, bc1;
#pragma unroll
  for (int e = 0; e < 16; ++e) {
    int K = amapK(e, half);
    bx0[e] = f2bf(xs[K * 16 + lm]);
    bx1[e] = f2bf(xs[(K + 32) * 16 + lm]);
    bc0[e] = f2bf(cs[K * 16 + lm]);
    bc1[e] = f2bf(cs[(K + 32) * 16 + lm]);
  }

  v8f qa = zero8(), ka = zero8(), va = zero8();
  qa = wmma_bf16(aq0, bx0, qa);  qa = wmma_bf16(aq1, bx1, qa);
  ka = wmma_bf16(ak0, bc0, ka);  ka = wmma_bf16(ak1, bc1, ka);
  va = wmma_bf16(av0, bc0, va);  va = wmma_bf16(av1, bc1, va);

  // ---- bias, stash q/k in LDS (fp32) for cross-wave RoPE pairing, store v
#pragma unroll
  for (int r = 0; r < 8; ++r) {
    int d = w * 16 + r + half * 8;
    float qv = qa[r] + bq[h * 64 + d];
    float kv = ka[r] + bk[h * 64 + d];
    float vv = va[r] + bv[h * 64 + d];
    qs[d * 16 + lm] = qv;
    ks[d * 16 + lm] = kv;
    vW[((size_t)bh * 64 + d) * T_ + t0 + lm] = f2bf(vv);
  }
  __syncthreads();

  // ---- RoPE (channels 0..31 rotated, 32..63 passthrough) and bf16 store
#pragma unroll
  for (int i = 0; i < 8; ++i) {
    int flat = tid + i * 128;   // 0..1023 over (d, n)
    int d = flat >> 4;
    int n = flat & 15;
    int tg = t0 + n;
    float qv, kv;
    if (d < 32) {
      int fi = d & 15;  // half = ROT/2 = 16
      float ang = __expf(-(float)fi * 0.57564627324851149f) * (float)tg;
      float sa = __sinf(ang), ca = __cosf(ang);
      float q1 = qs[fi * 16 + n], q2 = qs[(fi + 16) * 16 + n];
      float k1 = ks[fi * 16 + n], k2 = ks[(fi + 16) * 16 + n];
      if (d < 16) { qv = q1 * ca - q2 * sa; kv = k1 * ca - k2 * sa; }
      else        { qv = q1 * sa + q2 * ca; kv = k1 * sa + k2 * ca; }
    } else {
      qv = qs[d * 16 + n];
      kv = ks[d * 16 + n];
    }
    qT[((size_t)bh * T_ + tg) * 64 + d] = f2bf(qv);
    kT[((size_t)bh * T_ + tg) * 64 + d] = f2bf(kv);
  }
}

// ---------------------------------------------------------------------------
// Kernel 2: per-row online softmax stats (max, sumexp) over s.
// grid (B*H, T/16), one wave per 16 q-rows.
// ---------------------------------------------------------------------------
__global__ __launch_bounds__(32) void attn_stats_kernel(
    const __bf16* __restrict__ qT, const __bf16* __restrict__ kT,
    const int* __restrict__ mask, float* __restrict__ mrow,
    float* __restrict__ lrow) {
  const int bh = blockIdx.x;
  const int b = bh >> 4;
  const int t0 = blockIdx.y * 16;
  const int lane = threadIdx.x & 31;
  const int half = lane >> 4;
  const int lm = lane & 15;

  // A operand: Q^T rows t0+lm (contiguous d per lane)
  const __bf16* qrow = qT + ((size_t)bh * T_ + t0 + lm) * 64;
  v16bf a0 = *(const v16bf*)(qrow + half * 16);
  v16bf a1 = *(const v16bf*)(qrow + 32 + half * 16);

  float m[8], l[8];
#pragma unroll
  for (int r = 0; r < 8; ++r) { m[r] = -3.0e38f; l[r] = 0.f; }

  for (int s0 = 0; s0 < T_; s0 += 16) {
    const __bf16* krow = kT + ((size_t)bh * T_ + s0 + lm) * 64;
    __builtin_prefetch(krow + 16 * 64, 0, 0);  // next s-tile stream
    v16bf b0 = *(const v16bf*)(krow + half * 16);
    v16bf b1 = *(const v16bf*)(krow + 32 + half * 16);
    v8f S = zero8();
    S = wmma_bf16(a0, b0, S);
    S = wmma_bf16(a1, b1, S);
    int mk = mask[(size_t)b * T_ + s0 + lm];
#pragma unroll
    for (int r = 0; r < 8; ++r) {
      float sv = S[r] * 0.125f;
      if (mk == 0) sv = -10000.f;
      float rx = sv;
      rx = fmaxf(rx, __shfl_xor(rx, 1, 16));
      rx = fmaxf(rx, __shfl_xor(rx, 2, 16));
      rx = fmaxf(rx, __shfl_xor(rx, 4, 16));
      rx = fmaxf(rx, __shfl_xor(rx, 8, 16));
      float mnew = fmaxf(m[r], rx);
      float p = __expf(sv - mnew);
      p += __shfl_xor(p, 1, 16);
      p += __shfl_xor(p, 2, 16);
      p += __shfl_xor(p, 4, 16);
      p += __shfl_xor(p, 8, 16);
      l[r] = l[r] * __expf(m[r] - mnew) + p;
      m[r] = mnew;
    }
  }
  if (lm == 0) {
#pragma unroll
    for (int r = 0; r < 8; ++r) {
      int t = t0 + r + half * 8;
      mrow[(size_t)bh * T_ + t] = m[r];
      lrow[(size_t)bh * T_ + t] = l[r];
    }
  }
}

// ---------------------------------------------------------------------------
// Kernel 3: out[c, s-tile] = sum_t V[c,t] * exp(S[t,s]-m_t)/l_t.
// grid (B*H, T/16 s-tiles), one wave. Recomputes S (flash-style),
// transposes P through LDS into B-operand layout.
// ---------------------------------------------------------------------------
__global__ __launch_bounds__(32) void attn_out_kernel(
    const __bf16* __restrict__ qT, const __bf16* __restrict__ kT,
    const __bf16* __restrict__ vW, const int* __restrict__ mask,
    const float* __restrict__ mrow, const float* __restrict__ lrow,
    __bf16* __restrict__ aout) {
  __shared__ __bf16 P[16 * 32];  // [s_local][t_local]

  const int bh = blockIdx.x;
  const int b = bh >> 4, h = bh & 15;
  const int s0 = blockIdx.y * 16;
  const int lane = threadIdx.x & 31;
  const int half = lane >> 4;
  const int lm = lane & 15;

  // B operand for S (K columns s0+lm): loop-invariant
  const __bf16* krow = kT + ((size_t)bh * T_ + s0 + lm) * 64;
  v16bf kb0 = *(const v16bf*)(krow + half * 16);
  v16bf kb1 = *(const v16bf*)(krow + 32 + half * 16);
  const int mk = mask[(size_t)b * T_ + s0 + lm];

  v8f oacc[4];
#pragma unroll
  for (int mi = 0; mi < 4; ++mi) oacc[mi] = zero8();

  for (int t0 = 0; t0 < T_; t0 += 32) {
#pragma unroll
    for (int tau = 0; tau < 2; ++tau) {
      const __bf16* qrow = qT + ((size_t)bh * T_ + t0 + tau * 16 + lm) * 64;
      v16bf a0 = *(const v16bf*)(qrow + half * 16);
      v16bf a1 = *(const v16bf*)(qrow + 32 + half * 16);
      v8f S = zero8();
      S = wmma_bf16(a0, kb0, S);
      S = wmma_bf16(a1, kb1, S);
#pragma unroll
      for (int r = 0; r < 8; ++r) {
        int trow = t0 + tau * 16 + r + half * 8;
        float mv = mrow[(size_t)bh * T_ + trow];
        float lv = lrow[(size_t)bh * T_ + trow];
        float sv = S[r] * 0.125f;
        if (mk == 0) sv = -10000.f;
        float p = __expf(sv - mv) / lv;
        P[lm * 32 + tau * 16 + r + half * 8] = f2bf(p);
      }
    }
    __syncthreads();
    // B operand for P.V: K = t_local, N = s_local (contiguous per lane)
    v16bf pb = *(const v16bf*)(&P[lm * 32 + half * 16]);
#pragma unroll
    for (int mi = 0; mi < 4; ++mi) {
      const __bf16* vrow = vW + ((size_t)bh * 64 + mi * 16 + lm) * T_ + t0;
      v16bf va = *(const v16bf*)(vrow + half * 16);
      oacc[mi] = wmma_bf16(va, pb, oacc[mi]);
    }
    __syncthreads();
  }

#pragma unroll
  for (int mi = 0; mi < 4; ++mi)
#pragma unroll
    for (int r = 0; r < 8; ++r) {
      int crow = mi * 16 + r + half * 8;
      aout[((size_t)b * C_ + h * 64 + crow) * T_ + s0 + lm] = f2bf(oacc[mi][r]);
    }
}

// ---------------------------------------------------------------------------
// Kernel 4: y = wo @ out + bo. grid (B, COUT/64, T/16), block 128 (4 waves).
// B tile (32 k-rows x 16 t-cols) staged global->LDS with the Tensor Data
// Mover (TENSOR_LOAD_TO_LDS, TENSORcnt), issued once per block by wave 0.
// ---------------------------------------------------------------------------
__global__ __launch_bounds__(128) void proj_kernel(
    const __bf16* __restrict__ wo_bf, const __bf16* __restrict__ aout,
    const float* __restrict__ bo, float* __restrict__ out) {
  __shared__ __bf16 Bt[32 * 16];  // [k][n], exactly the TDM packed tile order

  const int b = blockIdx.x;
  const int o0 = blockIdx.y * 64;
  const int t0 = blockIdx.z * 16;
  const int tid = threadIdx.x;
  const int lane = tid & 31;
  const int w = tid >> 5;
  const int half = lane >> 4;
  const int lm = lane & 15;

  const unsigned int ldsBase = (unsigned int)(size_t)(&Bt[0]);

  v8f acc = zero8();
  const __bf16* arow = wo_bf + (size_t)(o0 + w * 16 + lm) * C_;

  for (int k0 = 0; k0 < C_; k0 += 32) {
    if (w == 0) {
      tdm_load_tile_2d(ldsBase, aout + ((size_t)b * C_ + k0) * T_ + t0);
      __builtin_amdgcn_s_wait_tensorcnt(0);
    }
    __syncthreads();

    v16bf a = *(const v16bf*)(arow + k0 + half * 16);
    v16bf bm;
#pragma unroll
    for (int e = 0; e < 16; ++e) bm[e] = Bt[(e + half * 16) * 16 + lm];
    acc = wmma_bf16(a, bm, acc);
    __syncthreads();
  }

#pragma unroll
  for (int r = 0; r < 8; ++r) {
    int o = o0 + w * 16 + r + half * 8;
    out[((size_t)b * COUT_ + o) * T_ + t0 + lm] = acc[r] + bo[o];
  }
}

// ---------------------------------------------------------------------------
// Launch
// ---------------------------------------------------------------------------
extern "C" void kernel_launch(void* const* d_in, const int* in_sizes, int n_in,
                              void* d_out, int out_size, void* d_ws,
                              size_t ws_size, hipStream_t stream) {
  (void)in_sizes; (void)n_in; (void)out_size; (void)ws_size;
  const float* x  = (const float*)d_in[0];
  const float* cc = (const float*)d_in[1];
  const int* msk  = (const int*)d_in[2];
  const float* wq = (const float*)d_in[3];
  const float* bq = (const float*)d_in[4];
  const float* wk = (const float*)d_in[5];
  const float* bk = (const float*)d_in[6];
  const float* wv = (const float*)d_in[7];
  const float* bv = (const float*)d_in[8];
  const float* wo = (const float*)d_in[9];
  const float* bo = (const float*)d_in[10];
  float* y = (float*)d_out;

  char* ws = (char*)d_ws;
  const size_t QKV = (size_t)B_ * H_ * T_ * CK_ * sizeof(__bf16);  // 8 MB each
  __bf16* qT    = (__bf16*)(ws);
  __bf16* kT    = (__bf16*)(ws + QKV);
  __bf16* vW    = (__bf16*)(ws + 2 * QKV);
  __bf16* aout  = (__bf16*)(ws + 3 * QKV);
  __bf16* wo_bf = (__bf16*)(ws + 4 * QKV);
  float*  mrow  = (float*)(ws + 4 * QKV + (size_t)COUT_ * C_ * sizeof(__bf16));
  float*  lrow  = mrow + (size_t)B_ * H_ * T_;

  cvt_wo_kernel<<<dim3((COUT_ * C_) / 256), dim3(256), 0, stream>>>(wo, wo_bf);

  qkv_rope_kernel<<<dim3(B_ * H_, T_ / 16), dim3(128), 0, stream>>>(
      x, cc, wq, bq, wk, bk, wv, bv, qT, kT, vW);

  attn_stats_kernel<<<dim3(B_ * H_, T_ / 16), dim3(32), 0, stream>>>(
      qT, kT, msk, mrow, lrow);

  attn_out_kernel<<<dim3(B_ * H_, T_ / 16), dim3(32), 0, stream>>>(
      qT, kT, vW, msk, mrow, lrow, aout);

  proj_kernel<<<dim3(B_, COUT_ / 64, T_ / 16), dim3(128), 0, stream>>>(
      wo_bf, aout, bo, y);
}
